// DecoderLayer_52132313039489
// MI455X (gfx1250) — matrix operand order repaired and checked by reading, not compiled, not run on hardware
//
#include <hip/hip_runtime.h>
#include <math.h>

// ---------------------------------------------------------------------------
// DecoderLayer for MI455X (gfx1250, wave32, WMMA).
// All heavy math runs through v_wmma_f32_16x16x32_bf16 (fp32 -> bf16 at tile
// load, fp32 accumulation). Attention keeps the whole 64xLk score block in the
// 320KB WGP LDS. B=16, L=512, D=512, H=8, DK=64, DFF=2048, NPIX=196, AL=20.
// ---------------------------------------------------------------------------

typedef __bf16 bf16_t;
typedef __attribute__((ext_vector_type(16))) __bf16 v16bf;
typedef __attribute__((ext_vector_type(8)))  float  v8f;

__device__ __forceinline__ v8f wmma_bf16(v16bf a, v16bf b, v8f c) {
  // 8 args: (neg_a, A, neg_b, B, c_mod, C, reuse_a, reuse_b)
  return __builtin_amdgcn_wmma_f32_16x16x32_bf16(false, a, false, b, (short)0, c,
                                                 false, false);
}

// A-matrix (16x32 bf16) half-index -> K index, contiguous order (matches B).
// lanes 0-15: K = i; lanes 16-31: K = 16 + i.
__device__ __forceinline__ int a_k_of(int i, int hi) {
  return (hi ? 16 : 0) + i;
}

__device__ __forceinline__ float sigm(float x) { return 1.0f / (1.0f + __expf(-x)); }

// gk in [0, 2*kmod): cheap wrap without integer division.
__device__ __forceinline__ int kwrap(int gk, int kmod) {
  return (gk >= kmod) ? gk - kmod : gk;
}

// ---------------------------------------------------------------------------
// Generic GEMM: C[M,N] = epilogue(A[M,K] @ W[N,K]^T + bias) (+ resid)
// A element (m,k) is read from A[m*lda + kwrap(k, a_kmod)]  (a_kmod implements
// the concat([img,img]) input without materializing it; K <= 2*a_kmod always).
// ep: 0 = none, 1 = relu, 2 = sigmoid.
// Block tile 128x64, 8 waves as 4(M) x 2(N); BK=64 -> 8 WMMAs per K-step.
// Register-staged software pipeline: fetch K-step n+1 into VGPRs while WMMAs
// consume K-step n from LDS; prefetch tile n+2 into L2 (global_prefetch_b8).
// ---------------------------------------------------------------------------
#define BM 128
#define BN 64
#define BK 64

__global__ __launch_bounds__(256) void k_gemm(
    const float* __restrict__ A, int lda, int a_kmod,
    const float* __restrict__ Wt, const float* __restrict__ bias,
    const float* __restrict__ resid, float* __restrict__ C,
    int M, int N, int K, int ep)
{
  __shared__ bf16_t As[BM][BK + 8];
  __shared__ bf16_t Ws[BN][BK + 8];

  const int tid  = threadIdx.x;
  const int wave = tid >> 5;
  const int lane = tid & 31;
  const int hi   = (lane >= 16) ? 1 : 0;
  const int lmod = lane & 15;
  const int m0 = blockIdx.y * BM;
  const int n0 = blockIdx.x * BN;
  const int wm = wave >> 1;  // 0..3
  const int wn = wave & 1;   // 0..1

  // staging assignments (fixed per thread)
  const int arow = tid >> 1, akk0 = (tid & 1) * 32;  // A: 2 thr/row, 32 halves
  const int wrow = tid >> 2, wkk0 = (tid & 3) * 16;  // W: 4 thr/row, 16 halves
  const int agm = m0 + arow;
  const int wgn = n0 + wrow;

  float4 areg[8];  // 32 floats of A row-segment
  float4 wreg[4];  // 16 floats of W row-segment

  auto fetch = [&](int k0) {
    #pragma unroll
    for (int u = 0; u < 8; u++) {
      const int gk = k0 + akk0 + u * 4;
      float4 f = make_float4(0.f, 0.f, 0.f, 0.f);
      if (agm < M && gk + 3 < K) {
        f = *(const float4*)(A + (size_t)agm * lda + kwrap(gk, a_kmod));
      } else if (agm < M) {
        float v[4];
        #pragma unroll
        for (int t = 0; t < 4; t++) {
          const int g = gk + t;
          v[t] = (g < K) ? A[(size_t)agm * lda + kwrap(g, a_kmod)] : 0.0f;
        }
        f = make_float4(v[0], v[1], v[2], v[3]);
      }
      areg[u] = f;
    }
    #pragma unroll
    for (int u = 0; u < 4; u++) {
      const int gk = k0 + wkk0 + u * 4;
      float4 f = make_float4(0.f, 0.f, 0.f, 0.f);
      if (wgn < N && gk + 3 < K) {
        f = *(const float4*)(Wt + (size_t)wgn * K + gk);
      } else if (wgn < N) {
        float v[4];
        #pragma unroll
        for (int t = 0; t < 4; t++) {
          const int g = gk + t;
          v[t] = (g < K) ? Wt[(size_t)wgn * K + g] : 0.0f;
        }
        f = make_float4(v[0], v[1], v[2], v[3]);
      }
      wreg[u] = f;
    }
  };
  auto stage = [&]() {
    #pragma unroll
    for (int u = 0; u < 8; u++) {
      const int kk = akk0 + u * 4;
      As[arow][kk + 0] = (bf16_t)areg[u].x; As[arow][kk + 1] = (bf16_t)areg[u].y;
      As[arow][kk + 2] = (bf16_t)areg[u].z; As[arow][kk + 3] = (bf16_t)areg[u].w;
    }
    #pragma unroll
    for (int u = 0; u < 4; u++) {
      const int kk = wkk0 + u * 4;
      Ws[wrow][kk + 0] = (bf16_t)wreg[u].x; Ws[wrow][kk + 1] = (bf16_t)wreg[u].y;
      Ws[wrow][kk + 2] = (bf16_t)wreg[u].z; Ws[wrow][kk + 3] = (bf16_t)wreg[u].w;
    }
  };

  v8f acc[2][2];
  #pragma unroll
  for (int i = 0; i < 2; i++)
    #pragma unroll
    for (int j = 0; j < 2; j++)
      #pragma unroll
      for (int e = 0; e < 8; e++) acc[i][j][e] = 0.0f;

  // prologue: stage first K tile
  fetch(0);
  stage();
  __syncthreads();

  for (int k0 = 0; k0 < K; k0 += BK) {
    const bool more = (k0 + BK) < K;
    if (more) {
      fetch(k0 + BK);  // register-staged prefetch of next tile
      if (k0 + 2 * BK < K) {  // pre-warm L2 for tile after next
        __builtin_prefetch(A + (size_t)agm * lda + kwrap(k0 + 2 * BK + akk0, a_kmod), 0, 1);
        __builtin_prefetch(Wt + (size_t)wgn * K + k0 + 2 * BK + wkk0, 0, 1);
      }
    }

    // ---- 8 WMMAs from LDS ----
    #pragma unroll
    for (int kc = 0; kc < 2; kc++) {
      v16bf afr[2], bfr[2];
      #pragma unroll
      for (int i = 0; i < 2; i++) {
        const int m = wm * 32 + i * 16 + lmod;
        #pragma unroll
        for (int e = 0; e < 16; e++) afr[i][e] = As[m][kc * 32 + a_k_of(e, hi)];
      }
      #pragma unroll
      for (int j = 0; j < 2; j++) {
        const int n = wn * 32 + j * 16 + lmod;
        #pragma unroll
        for (int e = 0; e < 16; e++) bfr[j][e] = Ws[n][kc * 32 + (hi ? 16 : 0) + e];
      }
      #pragma unroll
      for (int i = 0; i < 2; i++)
        #pragma unroll
        for (int j = 0; j < 2; j++)
          acc[i][j] = wmma_bf16(afr[i], bfr[j], acc[i][j]);
    }

    __syncthreads();
    if (more) {
      stage();
      __syncthreads();
    }
  }

  // ---- epilogue + store (C layout: lane n, 8 rows per lane) ----
  #pragma unroll
  for (int i = 0; i < 2; i++) {
    #pragma unroll
    for (int j = 0; j < 2; j++) {
      const int nn = n0 + wn * 32 + j * 16 + lmod;
      if (nn >= N) continue;
      const float bv = bias ? bias[nn] : 0.0f;
      #pragma unroll
      for (int r = 0; r < 8; r++) {
        const int mm = m0 + wm * 32 + i * 16 + r + (hi ? 8 : 0);
        if (mm >= M) continue;
        float v = acc[i][j][r] + bv;
        if (ep == 1)      v = fmaxf(v, 0.0f);
        else if (ep == 2) v = sigm(v);
        if (resid) v += resid[(size_t)mm * N + nn];
        C[(size_t)mm * N + nn] = v;
      }
    }
  }
}

// ---------------------------------------------------------------------------
// Fused attention for one (b,h) head slice and a 64-row Q tile.
// Keeps 64xLkp scores (f32) + full bf16 K/V panels in LDS (<=289KB of the
// 320KB WGP LDS for Lk=512). QK^T and PV both via WMMA bf16.
// attn_out (optional): [B,H,Lq,Lk] f32 probabilities.
// ---------------------------------------------------------------------------
__global__ __launch_bounds__(256) void k_attn(
    const float* __restrict__ Qb, const float* __restrict__ Kb,
    const float* __restrict__ Vb, float* __restrict__ ctx,
    float* __restrict__ attn_out, int Lq, int Lk, int Lkp, int causal)
{
  extern __shared__ char smem[];
  const int DS = 72;  // padded d-stride (halves) for Q/K/V panels
  bf16_t* Qs = (bf16_t*)smem;                    // 64 * DS
  bf16_t* Ks = Qs + 64 * DS;                     // Lkp * DS
  bf16_t* Vs = Ks + (size_t)Lkp * DS;            // Lkp * DS
  float*  Ss = (float*)(Vs + (size_t)Lkp * DS);  // 64 * (Lkp+4)
  const int sstr = Lkp + 4;

  const int tid = threadIdx.x, wave = tid >> 5, lane = tid & 31;
  const int hi = (lane >= 16) ? 1 : 0, lmod = lane & 15;
  const int q0 = blockIdx.x * 64;
  const int bh = blockIdx.y;
  const int b = bh >> 3, h = bh & 7;  // H = 8
  const int D = 512;                  // H * DK

  // stage Q (64x64)
  {
    const int row = tid >> 2, c0 = (tid & 3) * 16;
    const float* src = Qb + ((size_t)b * Lq + q0 + row) * D + h * 64 + c0;
    #pragma unroll
    for (int u = 0; u < 16; u++) Qs[row * DS + c0 + u] = (bf16_t)src[u];
  }
  // stage K, V (Lkp x 64, zero-padded rows)
  {
    const int c0 = (tid & 3) * 16;
    for (int r = tid >> 2; r < Lkp; r += 64) {
      if (r < Lk) {
        const float* ks = Kb + ((size_t)b * Lk + r) * D + h * 64 + c0;
        const float* vs = Vb + ((size_t)b * Lk + r) * D + h * 64 + c0;
        #pragma unroll
        for (int u = 0; u < 16; u++) {
          Ks[r * DS + c0 + u] = (bf16_t)ks[u];
          Vs[r * DS + c0 + u] = (bf16_t)vs[u];
        }
      } else {
        #pragma unroll
        for (int u = 0; u < 16; u++) {
          Ks[r * DS + c0 + u] = (bf16_t)0.0f;
          Vs[r * DS + c0 + u] = (bf16_t)0.0f;
        }
      }
    }
  }
  __syncthreads();

  // ---- S = scale * Q @ K^T, masked ----
  const int ntiles = Lkp >> 4;
  for (int tIdx = wave; tIdx < 4 * ntiles; tIdx += 8) {
    const int ms = tIdx / ntiles, nt = tIdx % ntiles;
    v8f acc;
    #pragma unroll
    for (int e = 0; e < 8; e++) acc[e] = 0.0f;
    #pragma unroll
    for (int kc = 0; kc < 2; kc++) {  // DK = 64 = 2 * 32
      v16bf afr, bfr;
      const int m = ms * 16 + lmod;
      const int n = nt * 16 + lmod;
      #pragma unroll
      for (int e = 0; e < 16; e++) {
        afr[e] = Qs[m * DS + kc * 32 + a_k_of(e, hi)];
        bfr[e] = Ks[n * DS + kc * 32 + (hi ? 16 : 0) + e];  // B[k][n]=K[n][k]
      }
      acc = wmma_bf16(afr, bfr, acc);
    }
    #pragma unroll
    for (int r = 0; r < 8; r++) {
      const int ml = ms * 16 + r + (hi ? 8 : 0);
      const int ng = nt * 16 + lmod;
      float v = acc[r] * 0.125f;  // 1/sqrt(64)
      if (ng >= Lk || (causal && ng > (q0 + ml))) v = -1e9f;
      Ss[ml * sstr + ng] = v;
    }
  }
  __syncthreads();

  // ---- softmax rows (4 lanes per row) + emit probabilities ----
  {
    const int row = tid >> 2, sub = tid & 3;
    float mx = -3.0e38f;
    for (int c = sub; c < Lkp; c += 4) mx = fmaxf(mx, Ss[row * sstr + c]);
    mx = fmaxf(mx, __shfl_xor(mx, 1));
    mx = fmaxf(mx, __shfl_xor(mx, 2));
    float sum = 0.0f;
    for (int c = sub; c < Lkp; c += 4) {
      const float e = __expf(Ss[row * sstr + c] - mx);
      Ss[row * sstr + c] = e;
      sum += e;
    }
    sum += __shfl_xor(sum, 1);
    sum += __shfl_xor(sum, 2);
    const float inv = 1.0f / sum;
    for (int c = sub; c < Lkp; c += 4) {
      const float p = Ss[row * sstr + c] * inv;
      Ss[row * sstr + c] = p;
      if (attn_out && c < Lk)
        attn_out[((size_t)bh * Lq + q0 + row) * Lk + c] = p;
    }
  }
  __syncthreads();

  // ---- O = P @ V  (64 x 64, K = Lkp) ----
  for (int tIdx = wave; tIdx < 16; tIdx += 8) {
    const int ms = tIdx >> 2, ns = tIdx & 3;
    v8f acc;
    #pragma unroll
    for (int e = 0; e < 8; e++) acc[e] = 0.0f;
    for (int kc = 0; kc < (Lkp >> 5); kc++) {
      v16bf afr, bfr;
      const int m = ms * 16 + lmod;
      const int n = ns * 16 + lmod;
      #pragma unroll
      for (int e = 0; e < 16; e++) {
        afr[e] = (bf16_t)Ss[m * sstr + kc * 32 + a_k_of(e, hi)];
        const int k = kc * 32 + (hi ? 16 : 0) + e;
        bfr[e] = Vs[k * DS + n];
      }
      acc = wmma_bf16(afr, bfr, acc);
    }
    #pragma unroll
    for (int r = 0; r < 8; r++) {
      const int mg = q0 + ms * 16 + r + (hi ? 8 : 0);
      const int ng = ns * 16 + lmod;
      ctx[((size_t)b * Lq + mg) * D + h * 64 + ng] = acc[r];
    }
  }
}

// ---------------------------------------------------------------------------
// LayerNorm over last dim 512. custom=1 -> ddof=1 std, eps=1e-6 added to std;
// custom=0 -> plain LN (biased var, eps=1e-5 inside sqrt, gamma=1, beta=0).
// One block of 256 threads per row; in-place safe.
// ---------------------------------------------------------------------------
__global__ __launch_bounds__(256) void k_ln(
    const float* __restrict__ x, const float* __restrict__ gam,
    const float* __restrict__ bet, float* __restrict__ y, int custom)
{
  __shared__ float red[8];
  const int row = blockIdx.x, tid = threadIdx.x;
  const float* xr = x + (size_t)row * 512;
  const float v0 = xr[tid], v1 = xr[tid + 256];

  float s = v0 + v1;
  #pragma unroll
  for (int o = 16; o > 0; o >>= 1) s += __shfl_xor(s, o);
  if ((tid & 31) == 0) red[tid >> 5] = s;
  __syncthreads();
  float mean = 0.0f;
  #pragma unroll
  for (int i = 0; i < 8; i++) mean += red[i];
  mean *= (1.0f / 512.0f);
  __syncthreads();

  const float d0 = v0 - mean, d1 = v1 - mean;
  s = d0 * d0 + d1 * d1;
  #pragma unroll
  for (int o = 16; o > 0; o >>= 1) s += __shfl_xor(s, o);
  if ((tid & 31) == 0) red[tid >> 5] = s;
  __syncthreads();
  float ss = 0.0f;
  #pragma unroll
  for (int i = 0; i < 8; i++) ss += red[i];

  float inv;
  if (custom) inv = 1.0f / (sqrtf(ss * (1.0f / 511.0f)) + 1e-6f);
  else        inv = rsqrtf(ss * (1.0f / 512.0f) + 1e-5f);

  const float g0 = gam ? gam[tid] : 1.0f, g1 = gam ? gam[tid + 256] : 1.0f;
  const float b0 = bet ? bet[tid] : 0.0f, b1 = bet ? bet[tid + 256] : 0.0f;
  y[(size_t)row * 512 + tid]       = g0 * d0 * inv + b0;
  y[(size_t)row * 512 + tid + 256] = g1 * d1 * inv + b1;
}

// GLU: [rows,512] -> [rows,256] : a * sigmoid(b)
__global__ void k_glu(const float* __restrict__ x, float* __restrict__ o, int rows) {
  const int i = blockIdx.x * 256 + threadIdx.x;
  if (i >= rows * 256) return;
  const int r = i >> 8, j = i & 255;
  const float a = x[(size_t)r * 512 + j];
  const float b = x[(size_t)r * 512 + 256 + j];
  o[i] = a * sigm(b);
}

// LSTM gate combine for timestep t (HG==nullptr means t==0: h=c=0 prior).
// G = x@W_ih^T + b_ih precomputed [B*L, 2048]; padded rows use b_ih only.
__global__ void k_lstm_gate(const float* __restrict__ G, const float* __restrict__ HG,
                            const float* __restrict__ b_ih, const float* __restrict__ b_hh,
                            float* __restrict__ c, float* __restrict__ h, int t)
{
  const int idx = blockIdx.x * blockDim.x + threadIdx.x;
  if (idx >= 16 * 512 * 512) return;
  const int j = idx & 511;
  const int pos = idx >> 9;      // b*L + l
  const int l = pos & 511;
  const int shift = 2 - t;       // KS-1-t
  float g[4];
  #pragma unroll
  for (int gi = 0; gi < 4; gi++) {
    const int col = gi * 512 + j;
    const float gx = (l >= shift) ? G[(size_t)(pos - shift) * 2048 + col] : b_ih[col];
    const float hh = HG ? HG[(size_t)pos * 2048 + col] : 0.0f;
    g[gi] = gx + hh + b_hh[col];
  }
  const float cp = HG ? c[idx] : 0.0f;
  const float cn = sigm(g[1]) * cp + sigm(g[0]) * tanhf(g[2]);
  c[idx] = cn;
  h[idx] = sigm(g[3]) * tanhf(cn);
}

__global__ void k_add(const float* __restrict__ a, const float* __restrict__ b,
                      float* __restrict__ o, int n) {
  const int i = blockIdx.x * 256 + threadIdx.x;
  if (i < n) o[i] = a[i] + b[i];
}

// y0 = ct * relu(img) + (1-ct) * relu(attr)
__global__ void k_mix(const float* __restrict__ ct, const float* __restrict__ img,
                      const float* __restrict__ at, float* __restrict__ y, int n) {
  const int i = blockIdx.x * 256 + threadIdx.x;
  if (i >= n) return;
  const float cv = ct[i];
  y[i] = cv * fmaxf(img[i], 0.0f) + (1.0f - cv) * fmaxf(at[i], 0.0f);
}

// ---------------------------------------------------------------------------
// Host orchestration
// ---------------------------------------------------------------------------
extern "C" void kernel_launch(void* const* d_in, const int* in_sizes, int n_in,
                              void* d_out, int out_size, void* d_ws, size_t ws_size,
                              hipStream_t stream) {
  (void)in_sizes; (void)n_in; (void)out_size; (void)ws_size;
  auto in = [&](int i) { return (const float*)d_in[i]; };

  const float* dec     = in(0);
  const float* enc     = in(1);
  const float* attr_in = in(2);
  // masks (3,4,5) are statically known: self=causal, enc/attr=all-false.
  const float* rnn_a = in(6),  * rnn_b = in(7);
  const float* conv_w = in(8), * conv_b = in(9);
  const float* w_ih = in(10), * w_hh = in(11), * b_ih = in(12), * b_hh = in(13);
  const float* sa_wq = in(14), * sa_bq = in(15), * sa_wk = in(16), * sa_bk = in(17);
  const float* sa_wv = in(18), * sa_bv = in(19), * sa_wo = in(20), * sa_bo = in(21);
  const float* ea_wq = in(22), * ea_bq = in(23), * ea_wk = in(24), * ea_bk = in(25);
  const float* ea_wv = in(26), * ea_bv = in(27), * ea_wo = in(28), * ea_bo = in(29);
  const float* aa_wq = in(30), * aa_bq = in(31), * aa_wk = in(32), * aa_bk = in(33);
  const float* aa_wv = in(34), * aa_bv = in(35), * aa_wo = in(36), * aa_bo = in(37);
  const float* fc_w = in(38), * fc_b = in(39);
  const float* norm_a = in(40), * norm_b = in(41);
  const float* ffn_w1 = in(42), * ffn_b1 = in(43);
  const float* ffn_w2 = in(44), * ffn_b2 = in(45);

  char* wsb = (char*)d_ws;
  auto wsf = [&](size_t mb) { return (float*)(wsb + mb * 1024ull * 1024ull); };
  // Phase 1 (LocalRNN):                    Phase 2+ (reuse of phase-1 space):
  float* xln   = wsf(0);                    float* qb      = wsf(0);
  float* glu   = wsf(16);                   float* kb      = wsf(16);
  float* convx = wsf(24);                   float* vb      = wsf(32);
  float* Gg    = wsf(40);                   float* ctxb    = wsf(48);
  float* HG    = wsf(104);                  float* dec_out = wsf(64);
  float* hbuf  = wsf(168);                  float* img     = wsf(80);
  float* cbuf  = wsf(184);                  float* attro   = wsf(96);
  float* x1    = wsf(200);                  float* ctb     = wsf(112);
  float* y0    = wsf(128);
  float* y1    = wsf(144);
  float* ffh   = wsf(160);  // 64MB, after x1 is dead

  float* out_y  = (float*)d_out;
  float* out_sa = out_y + (size_t)16 * 512 * 512;
  float* out_ea = out_sa + (size_t)16 * 8 * 512 * 512;

  const int ROWS = 16 * 512;          // B*L = 8192
  const int NEL  = ROWS * 512;        // 4.19M

  auto gemm = [&](const float* A, int lda, int kmod, const float* W,
                  const float* bias, const float* resid, float* C,
                  int M, int N, int K, int ep) {
    dim3 grid((N + BN - 1) / BN, (M + BM - 1) / BM);
    k_gemm<<<grid, 256, 0, stream>>>(A, lda, kmod, W, bias, resid, C, M, N, K, ep);
  };
  auto attn = [&](const float* Q, const float* K, const float* V, float* ctx,
                  float* ao, int Lk, int Lkp, int causal) {
    size_t sh = (size_t)64 * 72 * 2 + 2 * (size_t)Lkp * 72 * 2 +
                (size_t)64 * (Lkp + 4) * 4;
    k_attn<<<dim3(512 / 64, 16 * 8), 256, sh, stream>>>(Q, K, V, ctx, ao,
                                                        512, Lk, Lkp, causal);
  };

  // ---- LocalRNNLayer: x1 = dec + local_rnn(ln_custom(dec)) ----
  k_ln<<<ROWS, 256, 0, stream>>>(dec, rnn_a, rnn_b, xln, 1);
  k_glu<<<(ROWS * 256 + 255) / 256, 256, 0, stream>>>(xln, glu, ROWS);
  gemm(glu, 256, 256, conv_w, conv_b, nullptr, convx, ROWS, 512, 256, 0);
  gemm(convx, 512, 512, w_ih, b_ih, nullptr, Gg, ROWS, 2048, 512, 0);
  k_lstm_gate<<<(NEL + 255) / 256, 256, 0, stream>>>(Gg, nullptr, b_ih, b_hh, cbuf, hbuf, 0);
  gemm(hbuf, 512, 512, w_hh, nullptr, nullptr, HG, ROWS, 2048, 512, 0);
  k_lstm_gate<<<(NEL + 255) / 256, 256, 0, stream>>>(Gg, HG, b_ih, b_hh, cbuf, hbuf, 1);
  gemm(hbuf, 512, 512, w_hh, nullptr, nullptr, HG, ROWS, 2048, 512, 0);
  k_lstm_gate<<<(NEL + 255) / 256, 256, 0, stream>>>(Gg, HG, b_ih, b_hh, cbuf, hbuf, 2);
  k_add<<<(NEL + 255) / 256, 256, 0, stream>>>(dec, hbuf, x1, NEL);

  // ---- self-attention (causal), attn probs -> out_sa ----
  gemm(x1, 512, 512, sa_wq, sa_bq, nullptr, qb, ROWS, 512, 512, 0);
  gemm(x1, 512, 512, sa_wk, sa_bk, nullptr, kb, ROWS, 512, 512, 0);
  gemm(x1, 512, 512, sa_wv, sa_bv, nullptr, vb, ROWS, 512, 512, 0);
  attn(qb, kb, vb, ctxb, out_sa, 512, 512, 1);
  gemm(ctxb, 512, 512, sa_wo, sa_bo, x1, dec_out, ROWS, 512, 512, 0);
  k_ln<<<ROWS, 256, 0, stream>>>(dec_out, nullptr, nullptr, dec_out, 0);

  // ---- enc cross-attention (Lk=196, K=4096), attn probs -> out_ea ----
  gemm(dec_out, 512, 512, ea_wq, ea_bq, nullptr, qb, ROWS, 512, 512, 0);
  gemm(enc, 4096, 4096, ea_wk, ea_bk, nullptr, kb, 16 * 196, 512, 4096, 0);
  gemm(enc, 4096, 4096, ea_wv, ea_bv, nullptr, vb, 16 * 196, 512, 4096, 0);
  attn(qb, kb, vb, ctxb, out_ea, 196, 256, 0);
  gemm(ctxb, 512, 512, ea_wo, ea_bo, dec_out, img, ROWS, 512, 512, 0);
  k_ln<<<ROWS, 256, 0, stream>>>(img, nullptr, nullptr, img, 0);

  // ---- attr cross-attention (Lk=20, K=300) ----
  gemm(dec_out, 512, 512, aa_wq, aa_bq, nullptr, qb, ROWS, 512, 512, 0);
  gemm(attr_in, 300, 300, aa_wk, aa_bk, nullptr, kb, 16 * 20, 512, 300, 0);
  gemm(attr_in, 300, 300, aa_wv, aa_bv, nullptr, vb, 16 * 20, 512, 300, 0);
  attn(qb, kb, vb, ctxb, nullptr, 20, 32, 0);
  gemm(ctxb, 512, 512, aa_wo, aa_bo, dec_out, attro, ROWS, 512, 512, 0);
  k_ln<<<ROWS, 256, 0, stream>>>(attro, nullptr, nullptr, attro, 0);

  // ---- gated fusion: ct = sigmoid(fc([img,img])) via a_kmod=512 ----
  gemm(img, 512, 512, fc_w, fc_b, nullptr, ctb, ROWS, 512, 1024, 2);
  k_mix<<<(NEL + 255) / 256, 256, 0, stream>>>(ctb, img, attro, y0, NEL);
  k_ln<<<ROWS, 256, 0, stream>>>(y0, norm_a, norm_b, y1, 1);

  // ---- FFN + residual + plain LN -> out_y ----
  gemm(y1, 512, 512, ffn_w1, ffn_b1, nullptr, ffh, ROWS, 2048, 512, 1);
  gemm(ffh, 2048, 2048, ffn_w2, ffn_b2, y1, qb, ROWS, 512, 2048, 0);
  k_ln<<<ROWS, 256, 0, stream>>>(qb, nullptr, nullptr, out_y, 0);
}
